// GraphContrastiveModel_10866267258979
// MI455X (gfx1250) — matrix operand order repaired and checked by reading, not compiled, not run on hardware
//
#include <hip/hip_runtime.h>
#include <hip/hip_bf16.h>

// ---------------------------------------------------------------------------
// GCN contrastive loss for MI455X (gfx1250, wave32, WMMA).
// - GEMMs: v_wmma_f32_16x16x32_bf16 (bf16 in, f32 accum), wave-per-16x16-tile.
// - sim @ vg^T fused with online (flash) log-softmax; vg strips staged in LDS
//   via gfx1250 async-to-LDS copies (ASYNCcnt), double-buffered, shared by all
//   8 waves of the block (8x L2 traffic reduction).
// ---------------------------------------------------------------------------

typedef __bf16 bf16_t;
typedef __attribute__((ext_vector_type(16))) __bf16 v16bf;
typedef __attribute__((ext_vector_type(8)))  __bf16 v8bf;
typedef __attribute__((ext_vector_type(8)))  float  v8f;

union ABf16 { v16bf v; v8bf h[2]; };

// gfx1250 async global->LDS copy (16B per lane), tracked by ASYNCcnt.
__device__ __forceinline__ void async_b128(unsigned lds_off, const bf16_t* g) {
    asm volatile("global_load_async_to_lds_b128 %0, %1, off"
                 :: "v"(lds_off), "v"(g)
                 : "memory");
}

// ---------------- elementwise helpers ----------------

__global__ void k_fill_f32(float* __restrict__ p, float v, int n) {
    int i = blockIdx.x * blockDim.x + threadIdx.x;
    if (i < n) p[i] = v;
}

__global__ void k_deg(const int* __restrict__ dst, float* __restrict__ deg, int E) {
    int i = blockIdx.x * blockDim.x + threadIdx.x;
    if (i < E) atomicAdd(&deg[dst[i]], 1.0f);
}

__global__ void k_rsqrt(float* __restrict__ p, int n) {
    int i = blockIdx.x * blockDim.x + threadIdx.x;
    if (i < n) p[i] = rsqrtf(p[i]);
}

__global__ void k_f32_to_bf16(const float* __restrict__ in, bf16_t* __restrict__ out, int n) {
    int i = blockIdx.x * blockDim.x + threadIdx.x;
    if (i < n) out[i] = (bf16_t)in[i];
}

// Wt[f*K + k] = (bf16) W[k*F + f]   (transpose so WMMA B loads are contiguous)
__global__ void k_wt_bf16(const float* __restrict__ W, bf16_t* __restrict__ Wt, int K, int F) {
    int gid = blockIdx.x * blockDim.x + threadIdx.x;
    if (gid < K * F) {
        int f = gid / K;
        int k = gid - f * K;
        Wt[gid] = (bf16_t)W[k * F + f];
    }
}

__global__ void k_bias_act(float* __restrict__ X, const float* __restrict__ b,
                           int n, int Fmask, int dorelu) {
    int i = blockIdx.x * blockDim.x + threadIdx.x;
    if (i < n) {
        float v = X[i] + b[i & Fmask];
        if (dorelu) v = fmaxf(v, 0.0f);
        X[i] = v;
    }
}

// ---------------- WMMA GEMM: C[M,F] = A[M,256] @ Bt[F,256]^T ----------------
__global__ void k_gemm_bf16_k256(const bf16_t* __restrict__ A, const bf16_t* __restrict__ Bt,
                                 float* __restrict__ C, int M, int F) {
    const int K = 256;
    int lane = threadIdx.x & 31;
    int wid  = (blockIdx.x * blockDim.x + threadIdx.x) >> 5;
    int tn    = F >> 4;
    int total = (M >> 4) * tn;
    if (wid >= total) return;                  // wave-uniform: EXEC stays all-ones
    int mt = wid / tn, nt = wid - mt * tn;
    int hlf = lane >> 4, r = lane & 15;

    const bf16_t* arow = A  + (size_t)(mt * 16 + r) * K;
    const bf16_t* bcol = Bt + (size_t)(nt * 16 + r) * K;

    v8f c = {};
#pragma unroll
    for (int kb = 0; kb < K; kb += 32) {
        ABf16 a;
        a.h[0] = *(const v8bf*)(arow + kb + 16 * hlf);
        a.h[1] = *(const v8bf*)(arow + kb + 16 * hlf + 8);
        v16bf b = *(const v16bf*)(bcol + kb + 16 * hlf);
        c = __builtin_amdgcn_wmma_f32_16x16x32_bf16(false, a.v, false, b,
                                                    (short)0, c, false, false);
    }
    float* crow = C + (size_t)(mt * 16 + 8 * hlf) * F + nt * 16 + r;
#pragma unroll
    for (int v = 0; v < 8; ++v) crow[(size_t)v * F] = c[v];
}

// ---------------- GCN aggregation: out[dst] += norm * h[src] ----------------
__global__ void k_aggregate(const float* __restrict__ H, float* __restrict__ OUT,
                            const int* __restrict__ src, const int* __restrict__ dst,
                            const float* __restrict__ dinv, int E, int Etot, int logF) {
    long long gid = (long long)blockIdx.x * blockDim.x + threadIdx.x;
    if (gid >= ((long long)Etot << logF)) return;
    int e = (int)(gid >> logF);
    int f = (int)(gid & ((1 << logF) - 1));
    int s, d;
    if (e < E) { s = src[e]; d = dst[e]; }
    else       { s = e - E; d = s; }            // self-loop
    float nrm = dinv[s] * dinv[d];
    atomicAdd(OUT + ((size_t)d << logF) + f, H[((size_t)s << logF) + f] * nrm);
}

// ---------------- row L2-normalize (128 feats) -> bf16 ----------------
__global__ void k_l2norm_bf16(const float* __restrict__ X, bf16_t* __restrict__ Y, int N) {
    int lane = threadIdx.x & 31;
    int wid  = (blockIdx.x * blockDim.x + threadIdx.x) >> 5;
    if (wid >= N) return;
    const float* row = X + (size_t)wid * 128;
    float x[4]; float ss = 0.0f;
#pragma unroll
    for (int j = 0; j < 4; ++j) { x[j] = row[lane + 32 * j]; ss += x[j] * x[j]; }
#pragma unroll
    for (int d = 1; d < 32; d <<= 1) ss += __shfl_xor(ss, d);
    float inv = 1.0f / fmaxf(sqrtf(ss), 1e-12f);
    bf16_t* orow = Y + (size_t)wid * 128;
#pragma unroll
    for (int j = 0; j < 4; ++j) orow[lane + 32 * j] = (bf16_t)(x[j] * inv);
}

// ---------------- fused sim + streaming log-softmax + diag ----------------
// Block = 8 waves; wave w owns rows [16*(8*blk+w), +16). All waves share a
// 64-column vg strip staged in LDS by async-to-LDS copies (double buffered).
// Per strip: 16 WMMAs into 4 accumulators sharing the hoisted A operand, then
// one amortized max/sum-exp butterfly per row over all 64 columns.
__global__ void __launch_bounds__(256) k_sim_loss(const bf16_t* __restrict__ TG,
                                                  const bf16_t* __restrict__ VG,
                                                  float* __restrict__ rloss, int N) {
    const int K = 128;
    __shared__ __align__(32) bf16_t sB[2][64 * 128];   // 2 x 16KB strips
    const int tid  = threadIdx.x;
    const int lane = tid & 31;
    const int wv   = tid >> 5;
    const int rt   = blockIdx.x * 8 + wv;              // row tile (16 rows)
    const int hlf  = lane >> 4, r = lane & 15;
    const int nstrip = N >> 6;                          // 64-col strips

    // hoisted A operand (this wave's 16 tg rows, K=128 -> 4 k-steps)
    const bf16_t* arow = TG + (size_t)(rt * 16 + r) * K;
    ABf16 a[4];
#pragma unroll
    for (int j = 0; j < 4; ++j) {
        a[j].h[0] = *(const v8bf*)(arow + 32 * j + 16 * hlf);
        a[j].h[1] = *(const v8bf*)(arow + 32 * j + 16 * hlf + 8);
    }

    float m_run[8], s_run[8], diag[8];
#pragma unroll
    for (int v = 0; v < 8; ++v) { m_run[v] = -__builtin_inff(); s_run[v] = 0.0f; diag[v] = 0.0f; }

    // prologue: async-copy strip 0 (16KB = 256 threads x 4 x b128)
    {
        unsigned l0 = (unsigned)(size_t)&sB[0][0];
        const bf16_t* g = VG;
#pragma unroll
        for (int p = 0; p < 4; ++p)
            async_b128(l0 + (unsigned)(p * 256 + tid) * 16, g + (size_t)(p * 256 + tid) * 8);
    }

    for (int s = 0; s < nstrip; ++s) {
        if (s + 1 < nstrip) {           // issue next strip, then wait for current
            unsigned l1 = (unsigned)(size_t)&sB[(s + 1) & 1][0];
            const bf16_t* g = VG + (size_t)(s + 1) * 64 * K;
#pragma unroll
            for (int p = 0; p < 4; ++p)
                async_b128(l1 + (unsigned)(p * 256 + tid) * 16, g + (size_t)(p * 256 + tid) * 8);
            asm volatile("s_wait_asynccnt 0x4" ::: "memory");   // oldest 4 (strip s) done
        } else {
            asm volatile("s_wait_asynccnt 0x0" ::: "memory");
        }
        __syncthreads();

        const bf16_t* sb = &sB[s & 1][0];
        v8f c0 = {}, c1 = {}, c2 = {}, c3 = {};
#pragma unroll
        for (int j = 0; j < 4; ++j) {
            const bf16_t* bk = sb + 32 * j + 16 * hlf + (size_t)r * K;
            v16bf b0 = *(const v16bf*)(bk);
            v16bf b1 = *(const v16bf*)(bk + 16 * K);
            v16bf b2 = *(const v16bf*)(bk + 32 * K);
            v16bf b3 = *(const v16bf*)(bk + 48 * K);
            c0 = __builtin_amdgcn_wmma_f32_16x16x32_bf16(false, a[j].v, false, b0, (short)0, c0, false, false);
            c1 = __builtin_amdgcn_wmma_f32_16x16x32_bf16(false, a[j].v, false, b1, (short)0, c1, false, false);
            c2 = __builtin_amdgcn_wmma_f32_16x16x32_bf16(false, a[j].v, false, b2, (short)0, c2, false, false);
            c3 = __builtin_amdgcn_wmma_f32_16x16x32_bf16(false, a[j].v, false, b3, (short)0, c3, false, false);
        }

        const bool dstrip = ((rt >> 2) == s);   // this strip holds our diagonal
        const int  dt     = rt & 3;
#pragma unroll
        for (int v = 0; v < 8; ++v) {
            float s0 = fminf(fmaxf(c0[v] * 10.0f, -50.0f), 50.0f);   // sim/TEMP, clipped
            float s1 = fminf(fmaxf(c1[v] * 10.0f, -50.0f), 50.0f);
            float s2 = fminf(fmaxf(c2[v] * 10.0f, -50.0f), 50.0f);
            float s3 = fminf(fmaxf(c3[v] * 10.0f, -50.0f), 50.0f);
            if (dstrip) {
                float dsv = (dt == 0) ? s0 : (dt == 1) ? s1 : (dt == 2) ? s2 : s3;
                diag[v] = __shfl(dsv, hlf ? (24 + v) : v);           // element N==M
            }
            float mx = fmaxf(fmaxf(s0, s1), fmaxf(s2, s3));
#pragma unroll
            for (int dd = 1; dd < 16; dd <<= 1) mx = fmaxf(mx, __shfl_xor(mx, dd));
            float nm = fmaxf(m_run[v], mx);
            float ex = __expf(s0 - nm) + __expf(s1 - nm) + __expf(s2 - nm) + __expf(s3 - nm);
#pragma unroll
            for (int dd = 1; dd < 16; dd <<= 1) ex += __shfl_xor(ex, dd);
            s_run[v] = s_run[v] * __expf(m_run[v] - nm) + ex;
            m_run[v] = nm;
        }
        __syncthreads();
    }

    if (r == 0) {                               // lanes 0 and 16 write 8 rows each
        float* out = rloss + rt * 16 + 8 * hlf;
#pragma unroll
        for (int v = 0; v < 8; ++v) out[v] = m_run[v] + logf(s_run[v]) - diag[v];
    }
}

__global__ void k_mean(const float* __restrict__ x, float* __restrict__ out, int n) {
    __shared__ float sm[256];
    float s = 0.0f;
    for (int i = threadIdx.x; i < n; i += 256) s += x[i];
    sm[threadIdx.x] = s;
    __syncthreads();
    for (int d = 128; d > 0; d >>= 1) {
        if ((int)threadIdx.x < d) sm[threadIdx.x] += sm[threadIdx.x + d];
        __syncthreads();
    }
    if (threadIdx.x == 0) out[0] = sm[0] / (float)n;
}

// ---------------------------------------------------------------------------

static inline int cdiv(long long a, int b) { return (int)((a + b - 1) / b); }

struct Bufs {
    float*  DINV; bf16_t* XB;  float* H1; float* A1; bf16_t* HB;
    float*  H2;   float*  A2;
};

static void run_branch(const float* x, const int* ei, int N, int E,
                       const bf16_t* W1T, const bf16_t* W2T,
                       const float* b1, const float* b2,
                       const Bufs& B, bf16_t* EMB, hipStream_t stream) {
    const int* src = ei;
    const int* dst = ei + E;
    int Etot = E + N;

    // degrees (self-loop contributes 1) -> dinv = rsqrt(deg)
    k_fill_f32<<<cdiv(N, 256), 256, 0, stream>>>(B.DINV, 1.0f, N);
    k_deg<<<cdiv(E, 256), 256, 0, stream>>>(dst, B.DINV, E);
    k_rsqrt<<<cdiv(N, 256), 256, 0, stream>>>(B.DINV, N);

    // layer 1: h = x @ W1 ; aggregate ; +b1 ; relu
    k_f32_to_bf16<<<cdiv((long long)N * 256, 256), 256, 0, stream>>>(x, B.XB, N * 256);
    {
        int total = (N >> 4) * (256 >> 4);
        k_gemm_bf16_k256<<<cdiv((long long)total * 32, 256), 256, 0, stream>>>(B.XB, W1T, B.H1, N, 256);
    }
    hipMemsetAsync(B.A1, 0, (size_t)N * 256 * sizeof(float), stream);
    k_aggregate<<<cdiv((long long)Etot << 8, 256), 256, 0, stream>>>(B.H1, B.A1, src, dst, B.DINV, E, Etot, 8);
    k_bias_act<<<cdiv((long long)N * 256, 256), 256, 0, stream>>>(B.A1, b1, N * 256, 255, 1);

    // layer 2: h2 = a1 @ W2 ; aggregate ; +b2
    k_f32_to_bf16<<<cdiv((long long)N * 256, 256), 256, 0, stream>>>(B.A1, B.HB, N * 256);
    {
        int total = (N >> 4) * (128 >> 4);
        k_gemm_bf16_k256<<<cdiv((long long)total * 32, 256), 256, 0, stream>>>(B.HB, W2T, B.H2, N, 128);
    }
    hipMemsetAsync(B.A2, 0, (size_t)N * 128 * sizeof(float), stream);
    k_aggregate<<<cdiv((long long)Etot << 7, 256), 256, 0, stream>>>(B.H2, B.A2, src, dst, B.DINV, E, Etot, 7);
    k_bias_act<<<cdiv((long long)N * 128, 256), 256, 0, stream>>>(B.A2, b2, N * 128, 127, 0);

    // L2 normalize -> bf16 embedding
    k_l2norm_bf16<<<cdiv((long long)N * 32, 256), 256, 0, stream>>>(B.A2, EMB, N);
}

extern "C" void kernel_launch(void* const* d_in, const int* in_sizes, int n_in,
                              void* d_out, int out_size, void* d_ws, size_t ws_size,
                              hipStream_t stream) {
    const float* tg_x  = (const float*)d_in[0];
    const int*   tg_ei = (const int*)  d_in[1];
    const float* vg_x  = (const float*)d_in[2];
    const int*   vg_ei = (const int*)  d_in[3];
    const float* W1    = (const float*)d_in[4];
    const float* b1    = (const float*)d_in[5];
    const float* W2    = (const float*)d_in[6];
    const float* b2    = (const float*)d_in[7];

    const int N = in_sizes[0] / 256;   // 16384
    const int E = in_sizes[1] / 2;     // 524288

    // workspace carve-out (~76 MB, 256B-aligned slices)
    char* ws = (char*)d_ws;
    size_t off = 0;
    auto take = [&](size_t bytes) -> void* {
        void* p = ws + off;
        off += (bytes + 255) & ~(size_t)255;
        return p;
    };
    Bufs B;
    B.DINV = (float*) take((size_t)N * 4);
    B.XB   = (bf16_t*)take((size_t)N * 256 * 2);
    B.H1   = (float*) take((size_t)N * 256 * 4);
    B.A1   = (float*) take((size_t)N * 256 * 4);
    B.HB   = (bf16_t*)take((size_t)N * 256 * 2);
    B.H2   = (float*) take((size_t)N * 128 * 4);
    B.A2   = (float*) take((size_t)N * 128 * 4);
    bf16_t* W1T  = (bf16_t*)take(256 * 256 * 2);
    bf16_t* W2T  = (bf16_t*)take(128 * 256 * 2);
    bf16_t* TGB  = (bf16_t*)take((size_t)N * 128 * 2);
    bf16_t* VGB  = (bf16_t*)take((size_t)N * 128 * 2);
    float*  RL   = (float*) take((size_t)N * 4);

    // transposed bf16 weights (shared by both branches)
    k_wt_bf16<<<cdiv(256 * 256, 256), 256, 0, stream>>>(W1, W1T, 256, 256);
    k_wt_bf16<<<cdiv(256 * 128, 256), 256, 0, stream>>>(W2, W2T, 256, 128);

    run_branch(tg_x, tg_ei, N, E, W1T, W2T, b1, b2, B, TGB, stream);
    run_branch(vg_x, vg_ei, N, E, W1T, W2T, b1, b2, B, VGB, stream);

    // fused sim @ vg^T + clipped log-softmax diag loss (LDS-staged, async copies)
    k_sim_loss<<<N / 128, 256, 0, stream>>>(TGB, VGB, RL, N);
    k_mean<<<1, 256, 0, stream>>>(RL, (float*)d_out, N);

    (void)n_in; (void)out_size; (void)ws_size;
}
